// MultiHeadAttention_12996571038281
// MI455X (gfx1250) — matrix operand order repaired and checked by reading, not compiled, not run on hardware
//
#include <hip/hip_runtime.h>

// ---------------- CDNA5 WMMA types ----------------
typedef __attribute__((ext_vector_type(16))) __bf16 v16bf;
typedef __attribute__((ext_vector_type(8)))  float  v8f;
typedef __attribute__((ext_vector_type(4)))  unsigned int u32x4;
typedef __attribute__((ext_vector_type(8)))  int i32x8;
typedef __attribute__((ext_vector_type(4)))  int i32x4;

union FragBF {
  v16bf v;
  uint4 q[2];
};

__device__ __forceinline__ v8f wmma_bf16(v16bf a, v16bf b, v8f c) {
  // D = A(16x32 bf16) * B(32x16 bf16) + C(16x16 f32)
  return __builtin_amdgcn_wmma_f32_16x16x32_bf16(
      /*neg_a=*/false, a, /*neg_b=*/false, b,
      /*c_mod=*/(short)0, c, /*reuse_a=*/false, /*reuse_b=*/false);
}

// f32 -> bf16 (round-to-nearest-even), bit-twiddle to avoid __bf16 arithmetic
__device__ __forceinline__ unsigned short f2b(float f) {
  union { float f; unsigned u; } v; v.f = f;
  unsigned r = v.u + 0x7FFFu + ((v.u >> 16) & 1u);
  return (unsigned short)(r >> 16);
}

static constexpr int B_   = 4;
static constexpr int S_   = 1024;
static constexpr int D_   = 1024;
static constexpr int H_   = 16;
static constexpr int DEP_ = 64;
static constexpr int M_   = B_ * S_;   // 4096 rows for projection GEMMs

// ---------------- TDM: tensor_load_to_lds of a 2-D bf16 tile ----------------
// Loads tile_d1 rows x tile_d0 elements (row stride `stride0` elements) from
// global `gptr` into LDS at byte offset `lds_byte_addr` (rows packed
// contiguously: [tile_d1][tile_d0] bf16).
__device__ __forceinline__ void tdm_tile_load_bf16(
    unsigned lds_byte_addr, const unsigned short* gptr,
    unsigned tensor_d0, unsigned tensor_d1,
    unsigned tile_d0, unsigned tile_d1,
    unsigned long long stride0_elems) {
  unsigned long long ga = (unsigned long long)(size_t)gptr;
  u32x4 g0;
  g0[0] = 1u;                                              // count=1 (valid user D#)
  g0[1] = lds_byte_addr;                                   // bits 63:32 lds_addr
  g0[2] = (unsigned)(ga & 0xFFFFFFFFu);                    // global_addr[31:0]
  g0[3] = (unsigned)((ga >> 32) & 0x01FFFFFFu)             // global_addr[56:32]
          | 0x80000000u;                                   // type=2 ("image")

  union { unsigned w[8]; i32x8 v; } g1;
  g1.w[0] = 1u << 16;                                      // data_size=1 (2 bytes)
  g1.w[1] = (tensor_d0 & 0xFFFFu) << 16;                   // tensor_dim0[15:0] @ bits 63:48
  g1.w[2] = ((tensor_d0 >> 16) & 0xFFFFu)                  // tensor_dim0[31:16]
          | ((tensor_d1 & 0xFFFFu) << 16);                 // tensor_dim1[15:0]
  g1.w[3] = ((tensor_d1 >> 16) & 0xFFFFu)                  // tensor_dim1[31:16]
          | ((tile_d0 & 0xFFFFu) << 16);                   // tile_dim0
  g1.w[4] = (tile_d1 & 0xFFFFu);                           // tile_dim1 (tile_dim2=0)
  g1.w[5] = (unsigned)(stride0_elems & 0xFFFFFFFFu);       // tensor_dim0_stride[31:0]
  g1.w[6] = (unsigned)((stride0_elems >> 32) & 0xFFFFu);   // stride0[47:32]; stride1=0
  g1.w[7] = 0u;

  i32x4 z4 = {};
#if __has_include(<hip/amd_detail/amd_gfx1250_TDM.h>)
  i32x8 z8 = {};
  __builtin_amdgcn_tensor_load_to_lds(g0, g1.v, z4, z4, z8, 0);
#else
  __builtin_amdgcn_tensor_load_to_lds(g0, g1.v, z4, z4, 0);
#endif
}

// ---------------- elementwise f32 -> bf16 ----------------
__global__ void __launch_bounds__(256) cvt_f32_to_bf16(const float* __restrict__ in,
                                                       unsigned short* __restrict__ out,
                                                       int n4) {
  int i = blockIdx.x * blockDim.x + threadIdx.x;
  if (i < n4) {
    float4 f = ((const float4*)in)[i];
    uint2 o;
    o.x = (unsigned)f2b(f.x) | ((unsigned)f2b(f.y) << 16);
    o.y = (unsigned)f2b(f.z) | ((unsigned)f2b(f.w) << 16);
    ((uint2*)out)[i] = o;
  }
}

// ---------------- weight transpose + convert: W[K][N] f32 -> WT[N][K] bf16 ----
__global__ void __launch_bounds__(256) transpose_cvt(const float* __restrict__ W,
                                                     unsigned short* __restrict__ WT) {
  __shared__ float tile[32][33];
  const int tx = threadIdx.x & 31;
  const int ty = threadIdx.x >> 5;      // 0..7
  const int n0 = blockIdx.x * 32;
  const int k0 = blockIdx.y * 32;
#pragma unroll
  for (int i = 0; i < 4; ++i)
    tile[ty + i * 8][tx] = W[(size_t)(k0 + ty + i * 8) * D_ + n0 + tx];
  __syncthreads();
#pragma unroll
  for (int i = 0; i < 4; ++i)
    WT[(size_t)(n0 + ty + i * 8) * D_ + k0 + tx] = f2b(tile[tx][ty + i * 8]);
}

// ---------------- WMMA GEMM with TDM-staged LDS tiles ------------------------
// C[4096 x 1024] = A[4096 x 1024](bf16) @ WT^T + bias, WT stored [N][K] bf16.
// Block tile 128x128, 8 waves, each wave computes 32 rows x 64 cols.
// Per K-step (32 elems): TDM double-buffers A(128x32) and B(128x32) into LDS.
// MODE 0: store bf16 head-split [B,H,S,64]   (Q/K projections)
// MODE 1: store bf16 head-split transposed [B,H,64,S]  (V projection)
// MODE 2: store f32 [M,N]  (output projection -> d_out)
template <int MODE>
__global__ void __launch_bounds__(256) gemm_bf16_wmma(
    const unsigned short* __restrict__ A,
    const unsigned short* __restrict__ WT,
    const float* __restrict__ bias,
    void* __restrict__ out) {
  __shared__ unsigned short Abuf[2][128 * 32];   // 2 x 8 KB
  __shared__ unsigned short Bbuf[2][128 * 32];   // 2 x 8 KB

  const int wave = threadIdx.x >> 5;
  const int lane = threadIdx.x & 31;
  const int l16  = lane & 15;
  const int lh   = lane >> 4;
  const int rw   = wave >> 1;            // 0..3 : 32-row group
  const int cw   = wave & 1;             // 0..1 : 64-col group

  const int mBase = blockIdx.x * 128;
  const int nBase = blockIdx.y * 128;
  const unsigned short* Atile = A  + (size_t)mBase * D_;
  const unsigned short* Btile = WT + (size_t)nBase * D_;

  const unsigned aoff = (unsigned)(size_t)(&Abuf[0][0]);
  const unsigned boff = (unsigned)(size_t)(&Bbuf[0][0]);
  constexpr unsigned TILE_BYTES = 128 * 32 * 2;   // 8 KB

  const bool issuer = (wave == 0);
  if (issuer) {
    tdm_tile_load_bf16(aoff, Atile, D_, M_, 32, 128, D_);
    tdm_tile_load_bf16(boff, Btile, D_, D_, 32, 128, D_);
  }

  v8f acc[2][4] = {};
  for (int kt = 0; kt < D_ / 32; ++kt) {
    const int buf = kt & 1;
    if (issuer) {
      if (kt + 1 < D_ / 32) {
        const int nb = (kt + 1) & 1;
        tdm_tile_load_bf16(aoff + nb * TILE_BYTES, Atile + (kt + 1) * 32, D_, M_, 32, 128, D_);
        tdm_tile_load_bf16(boff + nb * TILE_BYTES, Btile + (kt + 1) * 32, D_, D_, 32, 128, D_);
        __builtin_amdgcn_s_wait_tensorcnt((short)2);   // current tile's 2 ops done
      } else {
        __builtin_amdgcn_s_wait_tensorcnt((short)0);
      }
    }
    __syncthreads();   // staged tile visible to all waves

    const unsigned short* Ab = &Abuf[buf][0];
    const unsigned short* Bb = &Bbuf[buf][0];
    FragBF afr[2];
#pragma unroll
    for (int rt = 0; rt < 2; ++rt) {
      const int row = rw * 32 + rt * 16 + l16;
      const unsigned short* ap = Ab + row * 32 + lh * 8;
      afr[rt].q[0] = *(const uint4*)(ap);
      afr[rt].q[1] = *(const uint4*)(ap + 16);
    }
#pragma unroll
    for (int c = 0; c < 4; ++c) {
      const int col = cw * 64 + c * 16 + l16;
      const unsigned short* bp = Bb + col * 32 + lh * 8;
      FragBF bfr;
      bfr.q[0] = *(const uint4*)(bp);
      bfr.q[1] = *(const uint4*)(bp + 16);
      acc[0][c] = wmma_bf16(afr[0].v, bfr.v, acc[0][c]);
      acc[1][c] = wmma_bf16(afr[1].v, bfr.v, acc[1][c]);
    }
    __syncthreads();   // protect buffer from next iteration's TDM overwrite
  }

#pragma unroll
  for (int rt = 0; rt < 2; ++rt) {
#pragma unroll
    for (int c = 0; c < 4; ++c) {
      const int n  = nBase + cw * 64 + c * 16 + l16;
      const float bv = bias[n];
#pragma unroll
      for (int r = 0; r < 8; ++r) {
        const int m = mBase + rw * 32 + rt * 16 + r + 8 * lh;  // C/D: M = r + 8*(lane/16)
        const float val = acc[rt][c][r] + bv;
        if (MODE == 0) {
          const int b = m >> 10, s = m & 1023, hh = n >> 6, d = n & 63;
          ((unsigned short*)out)[(((size_t)(b * H_ + hh) * S_ + s) * DEP_) + d] = f2b(val);
        } else if (MODE == 1) {
          const int b = m >> 10, s = m & 1023, hh = n >> 6, d = n & 63;
          ((unsigned short*)out)[(((size_t)(b * H_ + hh) * DEP_ + d) * S_) + s] = f2b(val);
        } else {
          ((float*)out)[(size_t)m * D_ + n] = val;
        }
      }
    }
  }
}

// ---------------- attention: scores + softmax + P@V, one (b,h) x 32 query rows / block
__global__ void __launch_bounds__(64) mha_attn_kernel(
    const unsigned short* __restrict__ Qh,   // [B,H,S,64] bf16
    const unsigned short* __restrict__ Kh,   // [B,H,S,64] bf16
    const unsigned short* __restrict__ Vt,   // [B,H,64,S] bf16
    unsigned short* __restrict__ attnOut) {  // [B,S,D] bf16 (heads merged)
  extern __shared__ char smem[];
  float* sc = (float*)smem;                                              // [32][1024] f32
  unsigned short* P = (unsigned short*)(smem + 32 * S_ * sizeof(float)); // [32][1024] bf16

  const int wave = threadIdx.x >> 5;
  const int lane = threadIdx.x & 31;
  const int l16  = lane & 15;
  const int lh   = lane >> 4;

  const int bh = blockIdx.y;              // b*H + h
  const int b  = bh >> 4;
  const int h  = bh & 15;
  const int s0 = blockIdx.x * 32 + wave * 16;   // first query row of this wave

  const unsigned short* Qb = Qh + (size_t)bh * S_ * DEP_;
  const unsigned short* Kb = Kh + (size_t)bh * S_ * DEP_;
  const unsigned short* Vb = Vt + (size_t)bh * DEP_ * S_;

  // Q A-fragments for the two depth k-steps; reused across all 64 key tiles
  FragBF qa[2];
#pragma unroll
  for (int kk = 0; kk < 2; ++kk) {
    const unsigned short* ap = Qb + (size_t)(s0 + l16) * DEP_ + kk * 32 + lh * 8;
    qa[kk].q[0] = *(const uint4*)(ap);
    qa[kk].q[1] = *(const uint4*)(ap + 16);
  }

  // ---- scores = (Q @ K^T) * 1/sqrt(64), 16x16 tiles over 64 key tiles ----
  for (int j = 0; j < 64; ++j) {
    v8f acc = {};
#pragma unroll
    for (int kk = 0; kk < 2; ++kk) {
      FragBF kf;   // B-fragment: column = key row j*16+l16, table-order depth
      const unsigned short* bp = Kb + (size_t)(j * 16 + l16) * DEP_ + kk * 32 + lh * 8;
      kf.q[0] = *(const uint4*)(bp);
      kf.q[1] = *(const uint4*)(bp + 16);
      acc = wmma_bf16(qa[kk].v, kf.v, acc);
    }
#pragma unroll
    for (int r = 0; r < 8; ++r)
      sc[(wave * 16 + r + 8 * lh) * S_ + j * 16 + l16] = acc[r] * 0.125f;
  }
  __syncthreads();

  // ---- row softmax on this wave's 16 rows (f32 in LDS -> bf16 P in LDS) ----
  for (int r = 0; r < 16; ++r) {
    float* row = sc + (wave * 16 + r) * S_;
    float mx = -3.402823e38f;
    for (int c = lane; c < S_; c += 32) mx = fmaxf(mx, row[c]);
#pragma unroll
    for (int off = 16; off > 0; off >>= 1) mx = fmaxf(mx, __shfl_xor(mx, off, 32));
    float sum = 0.f;
    for (int c = lane; c < S_; c += 32) {
      float e = __expf(row[c] - mx);
      row[c] = e;
      sum += e;
    }
#pragma unroll
    for (int off = 16; off > 0; off >>= 1) sum += __shfl_xor(sum, off, 32);
    const float inv = 1.0f / sum;
    unsigned short* prow = P + (wave * 16 + r) * S_;
    for (int c = lane; c < S_; c += 32) prow[c] = f2b(row[c] * inv);
  }
  __syncthreads();

  // ---- out(16x64) = P(16x1024) @ V^T ; A-fragments from LDS, B from L2 ----
  v8f oacc[4] = {};
  for (int kt = 0; kt < S_ / 32; ++kt) {
    const int kb2 = kt * 32;
    FragBF pa;
    const unsigned short* ap = P + (size_t)(wave * 16 + l16) * S_ + kb2 + lh * 8;
    pa.q[0] = *(const uint4*)(ap);
    pa.q[1] = *(const uint4*)(ap + 16);
#pragma unroll
    for (int c = 0; c < 4; ++c) {
      FragBF vf;
      const unsigned short* bp = Vb + (size_t)(c * 16 + l16) * S_ + kb2 + lh * 8;
      vf.q[0] = *(const uint4*)(bp);
      vf.q[1] = *(const uint4*)(bp + 16);
      oacc[c] = wmma_bf16(pa.v, vf.v, oacc[c]);
    }
  }
#pragma unroll
  for (int c = 0; c < 4; ++c) {
#pragma unroll
    for (int r = 0; r < 8; ++r) {
      const int sidx = s0 + r + 8 * lh;
      const int n    = h * DEP_ + c * 16 + l16;
      attnOut[(size_t)(b * S_ + sidx) * D_ + n] = f2b(oacc[c][r]);
    }
  }
}

// ---------------- host-side orchestration ----------------
extern "C" void kernel_launch(void* const* d_in, const int* in_sizes, int n_in,
                              void* d_out, int out_size, void* d_ws, size_t ws_size,
                              hipStream_t stream) {
  const float* v  = (const float*)d_in[0];
  const float* k  = (const float*)d_in[1];
  const float* q  = (const float*)d_in[2];
  const float* wq = (const float*)d_in[3];
  const float* bq = (const float*)d_in[4];
  const float* wk = (const float*)d_in[5];
  const float* bk = (const float*)d_in[6];
  const float* wv = (const float*)d_in[7];
  const float* bv = (const float*)d_in[8];
  const float* wo = (const float*)d_in[9];
  const float* bo = (const float*)d_in[10];

  unsigned short* ws = (unsigned short*)d_ws;
  const size_t ME = (size_t)M_ * D_;   // 4M elems per activation tensor
  const size_t WE = (size_t)D_ * D_;   // 1M elems per weight
  unsigned short* qb   = ws;
  unsigned short* kb   = ws + ME;
  unsigned short* vb   = ws + 2 * ME;
  unsigned short* wqT  = ws + 3 * ME;
  unsigned short* wkT  = wqT + WE;
  unsigned short* wvT  = wkT + WE;
  unsigned short* woT  = wvT + WE;
  unsigned short* Qh   = woT + WE;
  unsigned short* Kh   = Qh + ME;
  unsigned short* Vt   = Kh + ME;
  unsigned short* attn = Vt + ME;
  // total workspace use: 32M bf16 = 64 MB

  const int n4 = (int)(ME / 4);
  cvt_f32_to_bf16<<<n4 / 256, 256, 0, stream>>>(q, qb, n4);
  cvt_f32_to_bf16<<<n4 / 256, 256, 0, stream>>>(k, kb, n4);
  cvt_f32_to_bf16<<<n4 / 256, 256, 0, stream>>>(v, vb, n4);

  dim3 tgrid(D_ / 32, D_ / 32);
  transpose_cvt<<<tgrid, 256, 0, stream>>>(wq, wqT);
  transpose_cvt<<<tgrid, 256, 0, stream>>>(wk, wkT);
  transpose_cvt<<<tgrid, 256, 0, stream>>>(wv, wvT);
  transpose_cvt<<<tgrid, 256, 0, stream>>>(wo, woT);

  dim3 ggrid(M_ / 128, D_ / 128);
  gemm_bf16_wmma<0><<<ggrid, 256, 0, stream>>>(qb, wqT, bq, (void*)Qh);
  gemm_bf16_wmma<0><<<ggrid, 256, 0, stream>>>(kb, wkT, bk, (void*)Kh);
  gemm_bf16_wmma<1><<<ggrid, 256, 0, stream>>>(vb, wvT, bv, (void*)Vt);

  const size_t smem = 32 * (size_t)S_ * sizeof(float)
                    + 32 * (size_t)S_ * sizeof(unsigned short); // 192 KB (<320 KB/WGP)
  mha_attn_kernel<<<dim3(S_ / 32, B_ * H_), 64, smem, stream>>>(Qh, Kh, Vt, attn);

  gemm_bf16_wmma<2><<<ggrid, 256, 0, stream>>>(attn, woT, bo, d_out);
}
